// VanillaRNN_28501402976269
// MI455X (gfx1250) — matrix operand order repaired and checked by reading, not compiled, not run on hardware
//
#include <hip/hip_runtime.h>
#include <hip/hip_bf16.h>
#include <math.h>

// ---------------------------------------------------------------------------
// VanillaRNN forward for MI455X (gfx1250, wave32, WMMA bf16 16x16x32)
//   B=8, S=1024, H=512, V=32000
// Pipeline:
//   (a) convert Wxh/Whh/Wout -> transposed bf16 copies in workspace
//   (b) gather-convert embedding rows -> Abf [8192x512] bf16
//   (c) WMMA GEMM: xw = Abf @ Wxh          (f32 out, workspace)
//   (d) persistent 32-wave recurrence: hs = scan(tanh(xw_t + h@Whh + bh))
//   (e) WMMA GEMM: logits = hs @ Wout + bout  (dominant 268 GFLOP)
// Round 2 -> 3: GEMM restructured so 8 waves (8 M-tiles) share one B-tile
// stream staged through LDS via async global->LDS copies (ASYNCcnt path,
// triple-buffered, one barrier per tile). Fragment reads become ds_load_b128.
// ---------------------------------------------------------------------------

typedef __bf16  bf16_t;
typedef __bf16  v16bf __attribute__((ext_vector_type(16)));
typedef float   v8f   __attribute__((ext_vector_type(8)));

union Frag {            // one 16x32 (A) or 32x16 (B) bf16 WMMA operand: 8 VGPRs/lane
    v16bf v;
    uint4 q[2];
};

#define RNN_B   8
#define RNN_S   1024
#define RNN_H   512
#define RNN_V   32000
#define RNN_M   (RNN_B * RNN_S)        // 8192 rows
#define KCHUNKS (RNN_H / 32)           // 16 WMMA K-steps for K=512
#define TILE_BYTES (16 * RNN_H * 2)    // one B tile: 16 cols x K=512 bf16 = 16 KB

#if __has_builtin(__builtin_amdgcn_s_wait_asynccnt)
#define WAIT_ASYNC(n) __builtin_amdgcn_s_wait_asynccnt(n)
#else
#define WAIT_ASYNC(n) asm volatile("s_wait_asynccnt %0" :: "n"(n) : "memory")
#endif

// 16KB linear global->LDS copy: 256 threads x 4 x b128, ASYNCcnt += 4 / wave.
__device__ __forceinline__ void async_copy_tile(const bf16_t* __restrict__ gsrc,
                                                bf16_t* lds, int tid)
{
    #pragma unroll
    for (int r = 0; r < 4; ++r) {
        const void* g = (const char*)gsrc + tid * 16 + r * 4096;
        unsigned     l = (unsigned)(uintptr_t)((char*)lds + tid * 16 + r * 4096);
        asm volatile("global_load_async_to_lds_b128 %0, %1, off"
                     :: "v"(l), "v"(g) : "memory");
    }
}

// ---------------------------------------------------------------------------
// Converters
// ---------------------------------------------------------------------------

// in[K][N] f32 -> out[N][K] bf16 (N-major so B-fragments are contiguous K-runs)
__global__ __launch_bounds__(256) void k_transpose_bf16(
    const float* __restrict__ in, bf16_t* __restrict__ out, int K, int N)
{
    long idx = (long)blockIdx.x * 256 + threadIdx.x;
    if (idx >= (long)K * N) return;
    int k = (int)(idx / N);
    int n = (int)(idx % N);
    out[(size_t)n * K + k] = (bf16_t)in[idx];
}

// Abf[m][k] = bf16(embedding[token[m]][k]),  m = b*S + s
__global__ __launch_bounds__(256) void k_embed_bf16(
    const int* __restrict__ tok, const float* __restrict__ emb,
    bf16_t* __restrict__ Abf)
{
    int idx = blockIdx.x * 256 + threadIdx.x;      // over 8192*512
    int m = idx >> 9;
    int k = idx & (RNN_H - 1);
    Abf[idx] = (bf16_t)emb[(size_t)tok[m] * RNN_H + k];
}

// ---------------------------------------------------------------------------
// WMMA GEMM: out[M][N] = A[M][512](bf16) x BT[N][512](bf16, N-major) (+ bias)
// Block = 8 waves = 8 consecutive M-tiles (A fragments register-resident per
// wave). The block walks `ntPerBlock` N-tiles; each 16KB B tile (contiguous in
// BT) is staged into LDS with async b128 copies, triple-buffered so the next
// tile's global latency hides behind the current tile's 16 WMMAs.
// ---------------------------------------------------------------------------
__global__ __launch_bounds__(256) void k_wmma_gemm(
    const bf16_t* __restrict__ A,
    const bf16_t* __restrict__ BT,
    const float*  __restrict__ bias,   // may be nullptr
    float*        __restrict__ out,
    int N, int ntPerBlock)
{
    __shared__ alignas(16) bf16_t btile[3][16 * RNN_H];   // 48 KB of LDS

    const int tid  = threadIdx.x;
    const int lane = tid & 31;
    const int wave = tid >> 5;
    const int half = lane >> 4;        // 0: lanes 0-15, 1: lanes 16-31
    const int l16  = lane & 15;
    const int mtile = blockIdx.x * 8 + wave;   // one M-tile per wave
    const int nt0   = blockIdx.y * ntPerBlock;

    // Resident A fragments: row m = mtile*16 + l16, K-runs per ISA layout.
    Frag afr[KCHUNKS];
    {
        const bf16_t* ap = A + (size_t)(mtile * 16 + l16) * RNN_H + half * 8;
        #pragma unroll
        for (int c = 0; c < KCHUNKS; ++c) {
            afr[c].q[0] = *(const uint4*)(ap + c * 32);
            afr[c].q[1] = *(const uint4*)(ap + c * 32 + 16);
        }
    }

    // Prologue: start async copy of tile 0.
    async_copy_tile(BT + (size_t)nt0 * 16 * RNN_H, &btile[0][0], tid);

    for (int j = 0; j < ntPerBlock; ++j) {
        // Kick off tile j+1 while tile j finishes/computes.
        if (j + 1 < ntPerBlock) {
            async_copy_tile(BT + (size_t)(nt0 + j + 1) * 16 * RNN_H,
                            &btile[(j + 1) % 3][0], tid);
            WAIT_ASYNC(4);             // tile j's 4 copies done (in-order), j+1 in flight
        } else {
            WAIT_ASYNC(0);
        }
        __syncthreads();               // all waves' slices of tile j visible in LDS

        const bf16_t* bp = &btile[j % 3][l16 * RNN_H + half * 8];

        Frag bfr[2];                   // depth-1 pipelined LDS B stream
        bfr[0].q[0] = *(const uint4*)(bp);
        bfr[0].q[1] = *(const uint4*)(bp + 16);

        v8f acc0 = {}, acc1 = {};
        #pragma unroll
        for (int c = 0; c < KCHUNKS; ++c) {
            const int nc = c + 1;
            if (nc < KCHUNKS) {
                bfr[nc & 1].q[0] = *(const uint4*)(bp + nc * 32);
                bfr[nc & 1].q[1] = *(const uint4*)(bp + nc * 32 + 16);
            }
            if (c & 1)
                acc1 = __builtin_amdgcn_wmma_f32_16x16x32_bf16(
                    false, afr[c].v, false, bfr[c & 1].v, (short)0, acc1, false, false);
            else
                acc0 = __builtin_amdgcn_wmma_f32_16x16x32_bf16(
                    false, afr[c].v, false, bfr[c & 1].v, (short)0, acc0, false, false);
        }
        v8f acc = acc0 + acc1;

        const int n = (nt0 + j) * 16 + l16;
        const float bv = bias ? bias[n] : 0.0f;
        float* op = out + (size_t)(mtile * 16 + half * 8) * N + n;
        #pragma unroll
        for (int r = 0; r < 8; ++r)    // C layout: VGPR r -> row r (+8 for hi half)
            op[(size_t)r * N] = acc[r] + bv;
        // No trailing barrier: triple buffering + next iteration's barrier
        // guarantees tile (j%3) is not overwritten before all waves finish it.
    }
}

// ---------------------------------------------------------------------------
// Recurrence: one persistent 1024-thread block (32 waves on one WGP).
// Wave w owns h columns [16w,16w+16). Whh B-fragments (16 chunks = 128 VGPRs)
// stay resident across all 1024 steps; h ping-pongs through LDS as bf16 in
// A-friendly row-major [16][512] (rows 8..15 stay zero since B=8).
// ---------------------------------------------------------------------------
__global__ __launch_bounds__(1024) void k_rnn_scan(
    const float*  __restrict__ xw,     // [8192][512] f32, row = b*S + s
    const bf16_t* __restrict__ WhhT,   // [512][512] bf16, N-major
    const float*  __restrict__ bh,     // [512]
    bf16_t*       __restrict__ hs)     // [8192][512] bf16 out
{
    __shared__ bf16_t hbuf[2][16 * RNN_H];   // 32 KB of the 320 KB LDS

    const int lane = threadIdx.x & 31;
    const int wave = threadIdx.x >> 5;       // 0..31
    const int half = lane >> 4;
    const int l16  = lane & 15;
    const int n    = wave * 16 + l16;        // this lane's output column

    for (int i = threadIdx.x; i < 16 * RNN_H; i += 1024) {
        hbuf[0][i] = (bf16_t)0.0f;
        hbuf[1][i] = (bf16_t)0.0f;
    }

    Frag bfr[KCHUNKS];                       // resident Whh column-slice
    {
        const bf16_t* bp = WhhT + (size_t)n * RNN_H + half * 8;
        #pragma unroll
        for (int c = 0; c < KCHUNKS; ++c) {
            bfr[c].q[0] = *(const uint4*)(bp + c * 32);
            bfr[c].q[1] = *(const uint4*)(bp + c * 32 + 16);
        }
    }
    const float bhv = bh[n];
    __syncthreads();

    int p = 0;
    for (int t = 0; t < RNN_S; ++t) {
        const bf16_t* ap = &hbuf[p][l16 * RNN_H + half * 8];

        Frag afr[2];                         // depth-1 pipelined LDS A stream
        afr[0].q[0] = *(const uint4*)(ap);
        afr[0].q[1] = *(const uint4*)(ap + 16);

        v8f acc0 = {}, acc1 = {};
        #pragma unroll
        for (int c = 0; c < KCHUNKS; ++c) {
            const int nc = c + 1;
            if (nc < KCHUNKS) {
                afr[nc & 1].q[0] = *(const uint4*)(ap + nc * 32);
                afr[nc & 1].q[1] = *(const uint4*)(ap + nc * 32 + 16);
            }
            if (c & 1)
                acc1 = __builtin_amdgcn_wmma_f32_16x16x32_bf16(
                    false, afr[c & 1].v, false, bfr[c].v, (short)0, acc1, false, false);
            else
                acc0 = __builtin_amdgcn_wmma_f32_16x16x32_bf16(
                    false, afr[c & 1].v, false, bfr[c].v, (short)0, acc0, false, false);
        }
        v8f acc = acc0 + acc1;

        if (half == 0) {                     // rows 0..7 (= batch) live in lanes 0-15
            #pragma unroll
            for (int r = 0; r < RNN_B; ++r) {
                const size_t row = (size_t)r * RNN_S + t;
                float v = tanhf(acc[r] + xw[row * RNN_H + n] + bhv);
                bf16_t hv = (bf16_t)v;
                hs[row * RNN_H + n] = hv;
                hbuf[p ^ 1][r * RNN_H + n] = hv;
            }
        }
        __syncthreads();                     // release next buffer to all waves
        p ^= 1;
    }
}

// ---------------------------------------------------------------------------
// Host launch
// ---------------------------------------------------------------------------
extern "C" void kernel_launch(void* const* d_in, const int* in_sizes, int n_in,
                              void* d_out, int out_size, void* d_ws, size_t ws_size,
                              hipStream_t stream)
{
    const int*   tok  = (const int*)  d_in[0];   // [B,S]
    const float* emb  = (const float*)d_in[1];   // [V,H]
    const float* Wxh  = (const float*)d_in[2];   // [H,H]
    const float* Whh  = (const float*)d_in[3];   // [H,H]
    const float* bh   = (const float*)d_in[4];   // [H]
    const float* Wout = (const float*)d_in[5];   // [H,V]
    const float* bout = (const float*)d_in[6];   // [V]
    float* logits = (float*)d_out;               // [B,S,V]

    // Workspace carve-up (all offsets 1KB-aligned)
    char* ws = (char*)d_ws;
    float*  xw    = (float*) (ws);                                  // 16 MB
    bf16_t* Abf   = (bf16_t*)(ws + 16777216);                       //  8 MB
    bf16_t* hsbf  = (bf16_t*)(ws + 16777216 + 8388608);             //  8 MB
    bf16_t* WoutT = (bf16_t*)(ws + 16777216 + 2 * 8388608);         // 32.75 MB
    bf16_t* WxhT  = (bf16_t*)(ws + 16777216 + 2 * 8388608 + 32768000);
    bf16_t* WhhT  = (bf16_t*)(ws + 16777216 + 2 * 8388608 + 32768000 + 524288);

    // (a) weight conversions (bf16, transposed to N-major)
    k_transpose_bf16<<<(RNN_H * RNN_H + 255) / 256, 256, 0, stream>>>(Wxh, WxhT, RNN_H, RNN_H);
    k_transpose_bf16<<<(RNN_H * RNN_H + 255) / 256, 256, 0, stream>>>(Whh, WhhT, RNN_H, RNN_H);
    k_transpose_bf16<<<((long)RNN_H * RNN_V + 255) / 256, 256, 0, stream>>>(Wout, WoutT, RNN_H, RNN_V);

    // (b) embedding gather -> bf16
    k_embed_bf16<<<(RNN_M * RNN_H) / 256, 256, 0, stream>>>(tok, emb, Abf);

    // (c) xw = Abf @ Wxh : 64 m-superblocks x 4 n-blocks, 8 n-tiles each
    k_wmma_gemm<<<dim3(RNN_M / 128, 4), 256, 0, stream>>>(Abf, WxhT, nullptr, xw, RNN_H, 8);

    // (d) sequential scan, single persistent workgroup
    k_rnn_scan<<<1, 1024, 0, stream>>>(xw, WhhT, bh, hsbf);

    // (e) logits = hs @ Wout + bout : 64 m-superblocks x 50 n-blocks x 40 tiles
    k_wmma_gemm<<<dim3(RNN_M / 128, 50), 256, 0, stream>>>(hsbf, WoutT, bout, logits, RNN_V, 40);
}
